// PointNet2SemSegSSG_77257871720538
// MI455X (gfx1250) — matrix operand order repaired and checked by reading, not compiled, not run on hardware
//
#include <hip/hip_runtime.h>
#include <hip/hip_bf16.h>

typedef __attribute__((ext_vector_type(16))) _Float16 v16h;
typedef __attribute__((ext_vector_type(8)))  _Float16 v8h;
typedef __attribute__((ext_vector_type(8)))  float    v8f;

#define EPS_BN 1e-5f

// ---------------------------------------------------------------------------
// WMMA GEMM:  Y[M,Op](f16) = X[M,Kp](f16) * W[Op,Kp]^T (f16) + bias
// One wave computes a 16xOT*16 output strip. 8 waves / block.
// A layout (16x32 f16): lanes 0-15: m=lane, K = {0..7, 16..23}
//                       lanes16-31: m=lane-16, K = {8..15, 24..31}
// B layout (32x16 f16): lanes 0-15: n=lane, K = {0..7, 16..23}
//                       lanes16-31: n=lane-16, K = {8..15, 24..31}
// D layout: column n = o0 + (lane&15); row m = m0 + e + 8*(lane>=16)
// ---------------------------------------------------------------------------
template<int OT>
__global__ __launch_bounds__(256) void k_gemm_wmma(
    const _Float16* __restrict__ X, const _Float16* __restrict__ Wt,
    const float* __restrict__ bias, _Float16* __restrict__ Y,
    int M, int Kp, int Op)
{
  const int lane = threadIdx.x & 31;
  const int wave = threadIdx.x >> 5;
  const int m0 = (blockIdx.x * 8 + wave) * 16;
  if (m0 >= M) return;                      // wave-uniform
  const int o0 = blockIdx.y * (16 * OT);
  const int r  = lane & 15;
  const int ah = (lane >> 4) << 3;          // 0 or 8
  const int bh = (lane >> 4) << 4;          // 0 or 16

  v8f acc[OT];
#pragma unroll
  for (int t = 0; t < OT; ++t) { v8f z = {}; acc[t] = z; }

  const _Float16* xrow = X + (size_t)(m0 + r) * Kp;
  for (int k = 0; k < Kp; k += 32) {
    v8h alo = *(const v8h*)(xrow + k + ah);
    v8h ahi = *(const v8h*)(xrow + k + 16 + ah);
    v16h a = __builtin_shufflevector(alo, ahi, 0,1,2,3,4,5,6,7,8,9,10,11,12,13,14,15);
#pragma unroll
    for (int t = 0; t < OT; ++t) {
      const _Float16* wrow = Wt + (size_t)(o0 + t * 16 + r) * Kp;
      v8h blo = *(const v8h*)(wrow + k + ah);
      v8h bhi = *(const v8h*)(wrow + k + 16 + ah);
      v16h bm = __builtin_shufflevector(blo, bhi, 0,1,2,3,4,5,6,7,8,9,10,11,12,13,14,15);
      acc[t] = __builtin_amdgcn_wmma_f32_16x16x32_f16(
          false, a, false, bm, (short)0, acc[t], false, false);
    }
  }

  const int rh = (lane >> 4) << 3;          // +8 rows for hi half
#pragma unroll
  for (int t = 0; t < OT; ++t) {
    const int oc = o0 + t * 16 + r;
    const float bv = bias[oc];
#pragma unroll
    for (int e = 0; e < 8; ++e) {
      float v = acc[t][e] + bv;
      Y[(size_t)(m0 + e + rh) * Op + oc] = (_Float16)v;
    }
  }
}

// ---------------------------------------------------------------------------
// Input split: x(B,9,N) -> xyz(B,N,3), pts(B,N,6)
// ---------------------------------------------------------------------------
__global__ void k_split_input(const float* __restrict__ x, float* __restrict__ xyz,
                              float* __restrict__ pts, int B, int N)
{
  size_t t = (size_t)blockIdx.x * blockDim.x + threadIdx.x;
  size_t total = (size_t)B * N * 9;
  if (t >= total) return;
  int b = (int)(t / ((size_t)N * 9));
  size_t rem = t % ((size_t)N * 9);
  int n = (int)(rem / 9), c = (int)(rem % 9);
  float v = x[(size_t)b * 9 * N + (size_t)c * N + n];
  if (c < 3) xyz[((size_t)b * N + n) * 3 + c] = v;
  else       pts[((size_t)b * N + n) * 6 + (c - 3)] = v;
}

// ---------------------------------------------------------------------------
// Farthest point sampling (sequential per batch, distance array in LDS)
// ---------------------------------------------------------------------------
__global__ __launch_bounds__(256) void k_fps(const float* __restrict__ xyz,
                                             int* __restrict__ out, int N, int np)
{
  __shared__ float dist[4096];
  __shared__ float rv[256];
  __shared__ int   ri[256];
  __shared__ float cen[3];
  __shared__ int   cur;
  const int b = blockIdx.x, tid = threadIdx.x;
  const float* P = xyz + (size_t)b * N * 3;
  for (int i = tid; i < N; i += 256) dist[i] = 1e10f;
  if (tid == 0) cur = 0;
  __syncthreads();
  for (int it = 0; it < np; ++it) {
    int c = cur;
    if (tid == 0) {
      out[(size_t)b * np + it] = c;
      cen[0] = P[c * 3]; cen[1] = P[c * 3 + 1]; cen[2] = P[c * 3 + 2];
    }
    __syncthreads();
    float bx = cen[0], by = cen[1], bz = cen[2];
    float bestV = -1.f; int bestI = 0;
    for (int i = tid; i < N; i += 256) {
      float dx = P[i * 3] - bx, dy = P[i * 3 + 1] - by, dz = P[i * 3 + 2] - bz;
      float d = dx * dx + dy * dy + dz * dz;
      float o = dist[i];
      d = d < o ? d : o;
      dist[i] = d;
      if (d > bestV) { bestV = d; bestI = i; }
    }
    rv[tid] = bestV; ri[tid] = bestI;
    __syncthreads();
    for (int s = 128; s > 0; s >>= 1) {
      if (tid < s) {
        if (rv[tid + s] > rv[tid] ||
            (rv[tid + s] == rv[tid] && ri[tid + s] < ri[tid])) {
          rv[tid] = rv[tid + s]; ri[tid] = ri[tid + s];
        }
      }
      __syncthreads();
    }
    if (tid == 0) cur = ri[0];
    __syncthreads();
  }
}

__global__ void k_gather_xyz(const float* __restrict__ xyz, const int* __restrict__ fidx,
                             float* __restrict__ nxyz, int N, int np, size_t total)
{
  size_t t = (size_t)blockIdx.x * blockDim.x + threadIdx.x;
  if (t >= total) return;
  int c = (int)(t % 3);
  size_t q = t / 3;
  int j = (int)(q % np), b = (int)(q / np);
  nxyz[t] = xyz[((size_t)b * N + fidx[(size_t)b * np + j]) * 3 + c];
}

// ---------------------------------------------------------------------------
// Ball query: first ns indices (ascending) with d2 <= r2; fill with first
// ---------------------------------------------------------------------------
__global__ void k_ball(const float* __restrict__ xyz, const float* __restrict__ qxyz,
                       int* __restrict__ idx, int N, int np, int ns, float r2)
{
  extern __shared__ float sx[];                 // N*3 floats
  const int b = blockIdx.y;
  const float* P = xyz + (size_t)b * N * 3;
  for (int i = threadIdx.x; i < N * 3; i += blockDim.x) sx[i] = P[i];
  __syncthreads();
  int j = blockIdx.x * blockDim.x + threadIdx.x;
  if (j >= np) return;
  const float* q = qxyz + ((size_t)b * np + j) * 3;
  float qx = q[0], qy = q[1], qz = q[2];
  int* row = idx + ((size_t)b * np + j) * ns;
  int cnt = 0, first = -1;
  for (int i = 0; i < N && cnt < ns; ++i) {
    float dx = sx[i * 3] - qx, dy = sx[i * 3 + 1] - qy, dz = sx[i * 3 + 2] - qz;
    if (dx * dx + dy * dy + dz * dz <= r2) {
      if (first < 0) first = i;
      row[cnt++] = i;
    }
  }
  if (first < 0) first = N - 1;
  for (; cnt < ns; ++cnt) row[cnt] = first;
}

// ---------------------------------------------------------------------------
// Grouping for SA: X[row, 0..2]=xyz[idx]-new_xyz, [3..3+Cf)=feats[idx], pad 0
// ---------------------------------------------------------------------------
__global__ void k_group_sa(const float* __restrict__ xyz, const float* __restrict__ feats,
                           int Cf, const float* __restrict__ nxyz, const int* __restrict__ idx,
                           _Float16* __restrict__ X, int N, int np, int ns, int Kp, size_t total)
{
  size_t t = (size_t)blockIdx.x * blockDim.x + threadIdx.x;
  if (t >= total) return;
  size_t row = t / Kp;
  int k = (int)(t % Kp);
  int s = (int)(row % ns);
  size_t q = row / ns;
  int j = (int)(q % np), b = (int)(q / np);
  int i = idx[((size_t)b * np + j) * ns + s];
  float v = 0.f;
  if (k < 3)            v = xyz[((size_t)b * N + i) * 3 + k] - nxyz[((size_t)b * np + j) * 3 + k];
  else if (k < 3 + Cf)  v = feats[((size_t)b * N + i) * Cf + (k - 3)];
  X[t] = (_Float16)v;
}

// ---------------------------------------------------------------------------
// 3-NN (for feature propagation) + normalized inverse-distance weights
// ---------------------------------------------------------------------------
__global__ void k_knn3(const float* __restrict__ xyz1, const float* __restrict__ xyz2,
                       int* __restrict__ nidx, float* __restrict__ nw, int n1, int n2)
{
  extern __shared__ float s2[];                 // n2*3 floats
  const int b = blockIdx.y;
  const float* P2 = xyz2 + (size_t)b * n2 * 3;
  for (int i = threadIdx.x; i < n2 * 3; i += blockDim.x) s2[i] = P2[i];
  __syncthreads();
  int j = blockIdx.x * blockDim.x + threadIdx.x;
  if (j >= n1) return;
  const float* q = xyz1 + ((size_t)b * n1 + j) * 3;
  float qx = q[0], qy = q[1], qz = q[2];
  float d0 = 1e30f, d1 = 1e30f, d2 = 1e30f;
  int i0 = 0, i1 = 0, i2 = 0;
  for (int i = 0; i < n2; ++i) {
    float dx = s2[i * 3] - qx, dy = s2[i * 3 + 1] - qy, dz = s2[i * 3 + 2] - qz;
    float d = dx * dx + dy * dy + dz * dz;
    if (d < d0)      { d2 = d1; i2 = i1; d1 = d0; i1 = i0; d0 = d; i0 = i; }
    else if (d < d1) { d2 = d1; i2 = i1; d1 = d; i1 = i; }
    else if (d < d2) { d2 = d; i2 = i; }
  }
  float w0 = 1.f / (d0 + 1e-8f), w1 = 1.f / (d1 + 1e-8f), w2 = 1.f / (d2 + 1e-8f);
  float inv = 1.f / (w0 + w1 + w2);
  size_t base = ((size_t)b * n1 + j) * 3;
  nidx[base] = i0; nidx[base + 1] = i1; nidx[base + 2] = i2;
  nw[base] = w0 * inv; nw[base + 1] = w1 * inv; nw[base + 2] = w2 * inv;
}

// ---------------------------------------------------------------------------
// FP concat: X[row, 0..C1)=points1, [C1..C1+C2)=interp(points2), pad 0
// ---------------------------------------------------------------------------
__global__ void k_concat_fp(const float* __restrict__ p1, int C1,
                            const float* __restrict__ p2, int C2,
                            const int* __restrict__ nidx, const float* __restrict__ nw,
                            _Float16* __restrict__ X, int n1, int n2, int Kp, size_t total)
{
  size_t t = (size_t)blockIdx.x * blockDim.x + threadIdx.x;
  if (t >= total) return;
  size_t row = t / Kp;
  int k = (int)(t % Kp);
  int j = (int)(row % n1), b = (int)(row / n1);
  float v = 0.f;
  if (k < C1) {
    v = p1[((size_t)b * n1 + j) * C1 + k];
  } else if (k < C1 + C2) {
    int c = k - C1;
    size_t base = ((size_t)b * n1 + j) * 3;
    v = nw[base]     * p2[((size_t)b * n2 + nidx[base])     * C2 + c]
      + nw[base + 1] * p2[((size_t)b * n2 + nidx[base + 1]) * C2 + c]
      + nw[base + 2] * p2[((size_t)b * n2 + nidx[base + 2]) * C2 + c];
  }
  X[t] = (_Float16)v;
}

// ---------------------------------------------------------------------------
// BatchNorm: stats (LDS ds_add_f32 atomics -> global), finalize, apply+ReLU
// ---------------------------------------------------------------------------
__global__ __launch_bounds__(256) void k_bn_stats(const _Float16* __restrict__ H,
                                                  float* __restrict__ sums,
                                                  size_t total, int C)
{
  __shared__ float ls[512];
  __shared__ float lq[512];
  for (int i = threadIdx.x; i < C; i += blockDim.x) { ls[i] = 0.f; lq[i] = 0.f; }
  __syncthreads();
  size_t stride = (size_t)gridDim.x * blockDim.x;
  for (size_t g = (size_t)blockIdx.x * blockDim.x + threadIdx.x; g < total; g += stride) {
    float v = (float)H[g];
    int c = (int)(g % (size_t)C);
    atomicAdd(&ls[c], v);
    atomicAdd(&lq[c], v * v);
  }
  __syncthreads();
  for (int i = threadIdx.x; i < C; i += blockDim.x) {
    atomicAdd(&sums[i], ls[i]);
    atomicAdd(&sums[C + i], lq[i]);
  }
}

__global__ void k_bn_finalize(const float* __restrict__ sums, const float* __restrict__ gamma,
                              const float* __restrict__ beta, float* __restrict__ scale,
                              float* __restrict__ shift, int C, float invM)
{
  int c = blockIdx.x * blockDim.x + threadIdx.x;
  if (c >= C) return;
  float m = sums[c] * invM;
  float var = sums[C + c] * invM - m * m;
  float s = gamma[c] * rsqrtf(var + EPS_BN);
  scale[c] = s;
  shift[c] = beta[c] - m * s;
}

__global__ void k_bn_apply(_Float16* __restrict__ H, const float* __restrict__ scale,
                           const float* __restrict__ shift, size_t total, int C)
{
  size_t t = (size_t)blockIdx.x * blockDim.x + threadIdx.x;
  if (t >= total) return;
  int c = (int)(t % (size_t)C);
  float v = (float)H[t] * scale[c] + shift[c];
  H[t] = (_Float16)fmaxf(v, 0.f);
}

// ---------------------------------------------------------------------------
// Max pool over nsample, weight/bias conversion, casts, misc
// ---------------------------------------------------------------------------
__global__ void k_maxpool(const _Float16* __restrict__ H, float* __restrict__ out,
                          size_t rows, int ns, int C)
{
  size_t t = (size_t)blockIdx.x * blockDim.x + threadIdx.x;
  if (t >= rows * (size_t)C) return;
  size_t j = t / C;
  int c = (int)(t % C);
  const _Float16* p = H + (j * (size_t)ns) * C + c;
  float m = -1e30f;
  for (int s = 0; s < ns; ++s) m = fmaxf(m, (float)p[(size_t)s * C]);
  out[t] = m;
}

__global__ void k_conv_w(const float* __restrict__ W, _Float16* __restrict__ Wp,
                         int O, int K, int Op, int Kp)
{
  size_t t = (size_t)blockIdx.x * blockDim.x + threadIdx.x;
  if (t >= (size_t)Op * Kp) return;
  int o = (int)(t / Kp), k = (int)(t % Kp);
  Wp[t] = (_Float16)((o < O && k < K) ? W[(size_t)o * K + k] : 0.f);
}

__global__ void k_conv_b(const float* __restrict__ b, float* __restrict__ bp, int O, int Op)
{
  int t = blockIdx.x * blockDim.x + threadIdx.x;
  if (t >= Op) return;
  bp[t] = (t < O) ? b[t] : 0.f;
}

__global__ void k_zero(float* p, int n)
{
  int t = blockIdx.x * blockDim.x + threadIdx.x;
  if (t < n) p[t] = 0.f;
}

__global__ void k_f32_to_f16(const float* __restrict__ s, _Float16* __restrict__ d, size_t n)
{
  size_t t = (size_t)blockIdx.x * blockDim.x + threadIdx.x;
  if (t < n) d[t] = (_Float16)s[t];
}

__global__ void k_f16_to_f32(const _Float16* __restrict__ s, float* __restrict__ d, size_t n)
{
  size_t t = (size_t)blockIdx.x * blockDim.x + threadIdx.x;
  if (t < n) d[t] = (float)s[t];
}

// logits f16 [B*N x Cp] -> out f32 (B, C, N)
__global__ void k_write_out(const _Float16* __restrict__ L, float* __restrict__ out,
                            int B, int N, int C, int Cp)
{
  size_t t = (size_t)blockIdx.x * blockDim.x + threadIdx.x;
  size_t total = (size_t)B * C * N;
  if (t >= total) return;
  int b = (int)(t / ((size_t)C * N));
  size_t rem = t % ((size_t)C * N);
  int c = (int)(rem / N), n = (int)(rem % N);
  out[t] = (float)L[((size_t)b * N + n) * Cp + c];
}

// ---------------------------------------------------------------------------
// Host orchestration
// ---------------------------------------------------------------------------
extern "C" void kernel_launch(void* const* d_in, const int* in_sizes, int n_in,
                              void* d_out, int out_size, void* d_ws, size_t ws_size,
                              hipStream_t stream)
{
  (void)in_sizes; (void)n_in; (void)out_size; (void)ws_size;
  const int B = 16, N0 = 4096;

  char* base = (char*)d_ws;
  size_t off = 0;
  auto alloc = [&](size_t bytes) -> void* {
    off = (off + 255) & ~(size_t)255;
    void* p = base + off;
    off += bytes;
    return p;
  };
  auto cdiv = [](size_t a, size_t b) -> unsigned { return (unsigned)((a + b - 1) / b); };

  const size_t BUFE = (size_t)524288 * 64;            // max M*Op elements
  _Float16* bufA = (_Float16*)alloc(BUFE * 2);
  _Float16* bufB = (_Float16*)alloc(BUFE * 2);
  float* xyz0 = (float*)alloc((size_t)B * N0 * 3 * 4);
  float* pts0 = (float*)alloc((size_t)B * N0 * 6 * 4);
  float* l1x = (float*)alloc((size_t)B * 1024 * 3 * 4);
  float* l2x = (float*)alloc((size_t)B * 256 * 3 * 4);
  float* l3x = (float*)alloc((size_t)B * 64 * 3 * 4);
  float* l4x = (float*)alloc((size_t)B * 16 * 3 * 4);
  float* l1p = (float*)alloc((size_t)B * 1024 * 128 * 4);
  float* l2p = (float*)alloc((size_t)B * 256 * 256 * 4);
  float* l3p = (float*)alloc((size_t)B * 64 * 256 * 4);
  float* l4p = (float*)alloc((size_t)B * 16 * 512 * 4);
  float* l0p = (float*)alloc((size_t)B * 4096 * 128 * 4);
  int* fpsIdx = (int*)alloc((size_t)B * 1024 * 4);
  int* ballIdx = (int*)alloc((size_t)B * 1024 * 32 * 4);
  int* nnIdx = (int*)alloc((size_t)B * 4096 * 3 * 4);
  float* nnW = (float*)alloc((size_t)B * 4096 * 3 * 4);
  float* bnSums = (float*)alloc(1024 * 4);

  // Linear (+ optional BN+ReLU) layer; Y written in next layer's X layout.
  auto runLinear = [&](int pidx, int K, int O, _Float16* Xin, _Float16* Yout,
                       int M, bool bn) {
    int Kp = (K + 31) & ~31;
    int Op = (O + 15) & ~15;
    const float* W = (const float*)d_in[pidx];
    const float* bias = (const float*)d_in[pidx + 1];
    const float* gam = bn ? (const float*)d_in[pidx + 2] : nullptr;
    const float* bet = bn ? (const float*)d_in[pidx + 3] : nullptr;
    _Float16* Wp = (_Float16*)alloc((size_t)Op * Kp * 2);
    float* bp = (float*)alloc((size_t)Op * 4);
    float* sc = (float*)alloc((size_t)Op * 4);
    float* sh = (float*)alloc((size_t)Op * 4);
    {
      size_t n = (size_t)Op * Kp;
      k_conv_w<<<cdiv(n, 256), 256, 0, stream>>>(W, Wp, O, K, Op, Kp);
      k_conv_b<<<cdiv((size_t)Op, 256), 256, 0, stream>>>(bias, bp, O, Op);
    }
    dim3 blk(256);
    if (Op >= 64)
      k_gemm_wmma<4><<<dim3(M / 128, Op / 64), blk, 0, stream>>>(Xin, Wp, bp, Yout, M, Kp, Op);
    else if (Op == 32)
      k_gemm_wmma<2><<<dim3(M / 128, 1), blk, 0, stream>>>(Xin, Wp, bp, Yout, M, Kp, Op);
    else
      k_gemm_wmma<1><<<dim3(M / 128, 1), blk, 0, stream>>>(Xin, Wp, bp, Yout, M, Kp, Op);
    if (bn) {
      size_t total = (size_t)M * Op;
      k_zero<<<cdiv((size_t)(2 * Op), 256), 256, 0, stream>>>(bnSums, 2 * Op);
      size_t want = (total + 255) / 256;
      unsigned sb = (unsigned)(want < 2048 ? want : 2048);
      k_bn_stats<<<sb, 256, 0, stream>>>(Yout, bnSums, total, Op);
      k_bn_finalize<<<cdiv((size_t)Op, 256), 256, 0, stream>>>(bnSums, gam, bet, sc, sh, Op,
                                                               1.f / (float)M);
      k_bn_apply<<<cdiv(total, 256), 256, 0, stream>>>(Yout, sc, sh, total, Op);
    }
  };

  // Set-abstraction stage
  auto runSA = [&](const float* xyz, const float* feats, int Cf, int N, int np, float r2,
                   float* nxyz, float* outp, int pbase, const int* chans, int nch) {
    const int ns = 32;
    k_fps<<<B, 256, 0, stream>>>(xyz, fpsIdx, N, np);
    {
      size_t n = (size_t)B * np * 3;
      k_gather_xyz<<<cdiv(n, 256), 256, 0, stream>>>(xyz, fpsIdx, nxyz, N, np, n);
    }
    k_ball<<<dim3(cdiv((size_t)np, 128), B), 128, (size_t)N * 3 * 4, stream>>>(
        xyz, nxyz, ballIdx, N, np, ns, r2);
    int K0 = 3 + Cf, Kp0 = (K0 + 31) & ~31;
    int M = B * np * ns;
    {
      size_t n = (size_t)M * Kp0;
      k_group_sa<<<cdiv(n, 256), 256, 0, stream>>>(xyz, feats, Cf, nxyz, ballIdx, bufA,
                                                   N, np, ns, Kp0, n);
    }
    _Float16 *src = bufA, *dst = bufB;
    int K = K0;
    for (int l = 0; l < nch; ++l) {
      runLinear(pbase + 4 * l, K, chans[l], src, dst, M, true);
      _Float16* tmp = src; src = dst; dst = tmp;
      K = chans[l];
    }
    {
      int C = chans[nch - 1];
      size_t n = (size_t)B * np * C;
      k_maxpool<<<cdiv(n, 256), 256, 0, stream>>>(src, outp, (size_t)B * np, ns, C);
    }
  };

  // Feature-propagation stage
  auto runFP = [&](const float* xyz1, int n1, const float* xyz2, int n2,
                   const float* p1, int C1, const float* p2, int C2,
                   float* outp, int pbase, const int* chans, int nch) {
    k_knn3<<<dim3(cdiv((size_t)n1, 128), B), 128, (size_t)n2 * 3 * 4, stream>>>(
        xyz1, xyz2, nnIdx, nnW, n1, n2);
    int K0 = C1 + C2, Kp0 = (K0 + 31) & ~31;
    int M = B * n1;
    {
      size_t n = (size_t)M * Kp0;
      k_concat_fp<<<cdiv(n, 256), 256, 0, stream>>>(p1, C1, p2, C2, nnIdx, nnW, bufA,
                                                    n1, n2, Kp0, n);
    }
    _Float16 *src = bufA, *dst = bufB;
    int K = K0;
    for (int l = 0; l < nch; ++l) {
      runLinear(pbase + 4 * l, K, chans[l], src, dst, M, true);
      _Float16* tmp = src; src = dst; dst = tmp;
      K = chans[l];
    }
    {
      size_t n = (size_t)M * chans[nch - 1];
      k_f16_to_f32<<<cdiv(n, 256), 256, 0, stream>>>(src, outp, n);
    }
  };

  // ---- pipeline ----
  {
    size_t n = (size_t)B * N0 * 9;
    k_split_input<<<cdiv(n, 256), 256, 0, stream>>>((const float*)d_in[0], xyz0, pts0, B, N0);
  }

  const int c1[3] = {32, 32, 64};
  runSA(xyz0, pts0, 6, 4096, 1024, 0.01f, l1x, l1p, 1, c1, 3);
  const int c2[3] = {64, 64, 128};
  runSA(l1x, l1p, 64, 1024, 256, 0.04f, l2x, l2p, 13, c2, 3);
  const int c3[3] = {128, 128, 256};
  runSA(l2x, l2p, 128, 256, 64, 0.16f, l3x, l3p, 25, c3, 3);
  const int c4[3] = {256, 256, 512};
  runSA(l3x, l3p, 256, 64, 16, 0.64f, l4x, l4p, 37, c4, 3);

  const int f4[2] = {256, 256};
  runFP(l3x, 64, l4x, 16, l3p, 256, l4p, 512, l3p, 49, f4, 2);
  const int f3[2] = {256, 256};
  runFP(l2x, 256, l3x, 64, l2p, 128, l3p, 256, l2p, 57, f3, 2);
  const int f2[2] = {256, 128};
  runFP(l1x, 1024, l2x, 256, l1p, 64, l2p, 256, l1p, 65, f2, 2);
  const int f1[3] = {128, 128, 128};
  runFP(xyz0, 4096, l1x, 1024, pts0, 6, l1p, 128, l0p, 73, f1, 3);

  // head: Linear(128)+BN+ReLU, Linear(13)
  {
    size_t n = (size_t)B * 4096 * 128;
    k_f32_to_f16<<<cdiv(n, 256), 256, 0, stream>>>(l0p, bufA, n);
  }
  runLinear(85, 128, 128, bufA, bufB, B * 4096, true);
  runLinear(89, 128, 13, bufB, bufA, B * 4096, false);
  {
    size_t n = (size_t)B * 13 * 4096;
    k_write_out<<<cdiv(n, 256), 256, 0, stream>>>(bufA, (float*)d_out, B, 4096, 13, 16);
  }
}
